// SelfAttention_2611340116353
// MI455X (gfx1250) — compile-verified, matrix operand order checked by reading, not run on hardware
//
#include <hip/hip_runtime.h>
#include <hip/hip_bf16.h>
#include <math.h>

// Problem constants (fixed by the reference)
#define BB 4
#define SS 2048
#define DD 1024
#define HH 16
#define DH 64
#define D3 3072

typedef __bf16 bf16_t;
typedef __attribute__((ext_vector_type(16))) __bf16 v16bf;
typedef __attribute__((ext_vector_type(8)))  float  v8f;
typedef __attribute__((ext_vector_type(4)))  unsigned int u32x4;
typedef __attribute__((ext_vector_type(4)))  __bf16 v4bf;
typedef int i32x4_vs __attribute__((vector_size(16)));   // matches builtin param

#if __has_builtin(__builtin_amdgcn_global_load_async_to_lds_b128)
#define USE_ASYNC_LDS 1
#endif

union FragBF { v16bf v; u32x4 q[2]; };

__device__ __forceinline__ v8f wmma_bf16(v16bf a, v16bf b, v8f c) {
  // D = A(16x32 bf16) * B(32x16 bf16) + C(16x16 f32)
  return __builtin_amdgcn_wmma_f32_16x16x32_bf16(
      /*neg_a=*/false, a, /*neg_b=*/false, b,
      /*c_mod=*/(short)0, c, /*reuse_a=*/false, /*reuse_b=*/false);
}

// 16-byte global -> LDS copy: async DMA path on gfx1250 when available.
__device__ __forceinline__ void copy16_to_lds(void* lds, const bf16_t* g) {
#if defined(USE_ASYNC_LDS)
  __builtin_amdgcn_global_load_async_to_lds_b128(
      (__attribute__((address_space(1))) i32x4_vs*)(g),
      (__attribute__((address_space(3))) i32x4_vs*)(lds),
      /*offset=*/0, /*cpol=*/0);
#else
  *(u32x4*)lds = *(const u32x4*)g;
#endif
}

__device__ __forceinline__ void wait_lds_fill() {
#if defined(USE_ASYNC_LDS)
#if __has_builtin(__builtin_amdgcn_s_wait_asynccnt)
  __builtin_amdgcn_s_wait_asynccnt(0);
#else
  asm volatile("s_wait_asynccnt 0" ::: "memory");
#endif
#endif
}

// ---------------------------------------------------------------------------
// f32 -> bf16 conversion (vectorized x4)
// ---------------------------------------------------------------------------
__global__ __launch_bounds__(256)
void cvt_f32_bf16_kernel(const float* __restrict__ s, bf16_t* __restrict__ d, int n4) {
  int i = blockIdx.x * blockDim.x + threadIdx.x;
  if (i < n4) {
    const float4 f = ((const float4*)s)[i];
    v4bf o = { (bf16_t)f.x, (bf16_t)f.y, (bf16_t)f.z, (bf16_t)f.w };
    ((v4bf*)d)[i] = o;
  }
}

// ---------------------------------------------------------------------------
// Tiled bf16 GEMM:  C[M,N] = A[M,K] * W[N,K]^T + bias[N]
// Block tile 128x128 (8 waves as 4x2, each wave 32x64), K-step 32.
// LDS tiles filled with global_load_async_to_lds_b128 (ASYNCcnt path).
// MODE 0: QKV projection -> scatter to Q[B,H,S,Dh] (pre-scaled by 1/sqrt(Dh)),
//         K[B,H,S,Dh], Vt[B,H,Dh,S]  (all bf16)
// MODE 1: plain f32 output [M,N]
// ---------------------------------------------------------------------------
template <int MODE>
__global__ __launch_bounds__(256)
void gemm_bf16_nt(const bf16_t* __restrict__ A,   // [M,K] row-major bf16
                  const bf16_t* __restrict__ W,   // [N,K] row-major bf16
                  const float*  __restrict__ bias,// [N] f32
                  float* __restrict__ Cout,       // MODE 1
                  bf16_t* __restrict__ Qb,
                  bf16_t* __restrict__ Kb,
                  bf16_t* __restrict__ Vtb,
                  int M, int N, int K)
{
  constexpr int BM = 128, BN = 128, BK = 32, LDP = BK + 8; // padded LDS rows
  __shared__ bf16_t As[BM][LDP];
  __shared__ bf16_t Bs[BN][LDP];

  const int tid  = threadIdx.x;
  const int lane = tid & 31;
  const int wave = tid >> 5;
  const int wm   = wave & 3;   // wave row (4): 32 rows each
  const int wn   = wave >> 2;  // wave col (2): 64 cols each
  const int m0   = blockIdx.x * BM;
  const int n0   = blockIdx.y * BN;
  const int l15  = lane & 15;
  const int koff = (lane >> 4) * 8;   // K-half select per WMMA operand layout

  v8f acc[2][4];
#pragma unroll
  for (int i = 0; i < 2; ++i)
#pragma unroll
    for (int j = 0; j < 4; ++j)
      acc[i][j] = (v8f){};

  const int steps = K / BK;
  for (int kk = 0; kk < steps; ++kk) {
    const int k0 = kk * BK;
    __syncthreads();
    // Cooperative LDS fill via async DMA:
    // A tile 128x32 (512 x 16B), W tile 128x32 (512 x 16B)
#pragma unroll
    for (int t = 0; t < 2; ++t) {
      int idx = tid + t * 256;
      int row = idx >> 2, c8 = (idx & 3) * 8;
      copy16_to_lds(&As[row][c8], A + (size_t)(m0 + row) * K + k0 + c8);
    }
#pragma unroll
    for (int t = 0; t < 2; ++t) {
      int idx = tid + t * 256;
      int row = idx >> 2, c8 = (idx & 3) * 8;
      copy16_to_lds(&Bs[row][c8], W + (size_t)(n0 + row) * K + k0 + c8);
    }
    wait_lds_fill();     // s_wait_asynccnt 0 (no-op on fallback path)
    __syncthreads();
    if (kk + 1 < steps) { // speculative prefetch of next K-tiles (global_prefetch_b8)
      __builtin_prefetch(A + (size_t)(m0 + (tid >> 1)) * K + k0 + BK, 0, 0);
      __builtin_prefetch(W + (size_t)(n0 + (tid >> 1)) * K + k0 + BK, 0, 0);
    }

    FragBF fa[2], fb[4];
#pragma unroll
    for (int i = 0; i < 2; ++i) {
      int row = wm * 32 + i * 16 + l15;
      fa[i].q[0] = *(const u32x4*)(&As[row][koff]);
      fa[i].q[1] = *(const u32x4*)(&As[row][16 + koff]);
    }
#pragma unroll
    for (int j = 0; j < 4; ++j) {
      int col = wn * 64 + j * 16 + l15;
      fb[j].q[0] = *(const u32x4*)(&Bs[col][koff]);
      fb[j].q[1] = *(const u32x4*)(&Bs[col][16 + koff]);
    }
#pragma unroll
    for (int i = 0; i < 2; ++i)
#pragma unroll
      for (int j = 0; j < 4; ++j)
        acc[i][j] = wmma_bf16(fa[i].v, fb[j].v, acc[i][j]);
  }

  // Epilogue. C-layout: reg r, lane -> (row = r + 8*(lane>>4), col = lane&15)
  const int rbase = (lane >> 4) * 8;
#pragma unroll
  for (int i = 0; i < 2; ++i)
#pragma unroll
    for (int j = 0; j < 4; ++j)
#pragma unroll
      for (int r = 0; r < 8; ++r) {
        int m = m0 + wm * 32 + i * 16 + r + rbase;
        int n = n0 + wn * 64 + j * 16 + l15;
        float v = acc[i][j][r] + bias[n];
        if (MODE == 0) {
          int b = m >> 11, s = m & (SS - 1);
          int sect = n >> 10, e = n & (DD - 1);
          int h = e >> 6, d = e & (DH - 1);
          size_t bh = (size_t)(b * HH + h);
          if (sect == 0)      Qb[(bh * SS + s) * DH + d] = (bf16_t)(v * 0.125f); // 1/sqrt(64)
          else if (sect == 1) Kb[(bh * SS + s) * DH + d] = (bf16_t)v;
          else                Vtb[(bh * DH + d) * SS + s] = (bf16_t)v;           // transposed V
        } else {
          Cout[(size_t)m * N + n] = v;
        }
      }
}

// ---------------------------------------------------------------------------
// Flash attention, one (b,h) per blockIdx.y, 128 q-rows per block,
// 16 q-rows per wave, 32-key inner tile, online softmax.
// Row-sum is tracked via an extra WMMA against an all-ones B operand
// (rescales by alpha exactly like the O tiles), so only the row-max
// needs cross-lane shuffles.
// att output: [B,S,D] bf16 with head d placed at h*64+d.
// ---------------------------------------------------------------------------
__global__ __launch_bounds__(256)
void flash_attn_kernel(const bf16_t* __restrict__ Qb,
                       const bf16_t* __restrict__ Kb,
                       const bf16_t* __restrict__ Vtb,
                       bf16_t* __restrict__ att)
{
  __shared__ bf16_t Plds[8][16][40];   // per-wave 16x32 P tile (padded)

  const int tid  = threadIdx.x;
  const int lane = tid & 31;
  const int wave = tid >> 5;
  const int l15  = lane & 15;
  const int half = lane >> 4;
  const int koff = half * 8;
  const int bh   = blockIdx.y;
  const int b    = bh >> 4, h = bh & 15;
  const int q0   = blockIdx.x * 128 + wave * 16;

  const bf16_t* Qp = Qb  + (size_t)bh * SS * DH;
  const bf16_t* Kp = Kb  + (size_t)bh * SS * DH;
  const bf16_t* Vp = Vtb + (size_t)bh * DH * SS;

  // Q fragments: A-operand layout, loaded once (Q pre-scaled by 1/sqrt(Dh))
  FragBF fq[2];
  {
    const bf16_t* qrow = Qp + (size_t)(q0 + l15) * DH;
#pragma unroll
    for (int f = 0; f < 2; ++f) {
      fq[f].q[0] = *(const u32x4*)(qrow + f * 32 + koff);
      fq[f].q[1] = *(const u32x4*)(qrow + f * 32 + 16 + koff);
    }
  }

  // all-ones B operand for row-sum tracking
  FragBF fone;
#pragma unroll
  for (int i = 0; i < 16; ++i) fone.v[i] = (bf16_t)1.0f;

  v8f o[4], oL;
#pragma unroll
  for (int t = 0; t < 4; ++t) o[t] = (v8f){};
  oL = (v8f){};
  float mrow[8];
#pragma unroll
  for (int r = 0; r < 8; ++r) mrow[r] = -__builtin_inff();

  for (int kb = 0; kb < SS; kb += 32) {
    // ---- issue ALL global loads for this tile up front (loads retire in
    //      order, so score WMMAs only wait on the K subset while V flies) ----
    FragBF fk[2][2];
#pragma unroll
    for (int t = 0; t < 2; ++t) {
      const bf16_t* krow = Kp + (size_t)(kb + t * 16 + l15) * DH;
      fk[t][0].q[0] = *(const u32x4*)(krow + koff);
      fk[t][0].q[1] = *(const u32x4*)(krow + 16 + koff);
      fk[t][1].q[0] = *(const u32x4*)(krow + 32 + koff);
      fk[t][1].q[1] = *(const u32x4*)(krow + 48 + koff);
    }
    FragBF fv[4];
#pragma unroll
    for (int t = 0; t < 4; ++t) {
      const bf16_t* vrow = Vp + (size_t)(t * 16 + l15) * SS + kb;
      fv[t].q[0] = *(const u32x4*)(vrow + koff);
      fv[t].q[1] = *(const u32x4*)(vrow + 16 + koff);
    }
    if (kb + 32 < SS) { // prefetch next key tile (global_prefetch_b8)
      __builtin_prefetch(Kp + (size_t)(kb + 32 + lane) * DH, 0, 0);
      __builtin_prefetch(Vp + (size_t)lane * SS + kb + 32, 0, 0);
      __builtin_prefetch(Vp + (size_t)(lane + 32) * SS + kb + 32, 0, 0);
    }

    // ---- scores: S[16 q x 32 keys] via 4 WMMAs ----
    v8f sc[2];
#pragma unroll
    for (int t = 0; t < 2; ++t) {
      v8f z = (v8f){};
      z     = wmma_bf16(fq[0].v, fk[t][0].v, z);
      sc[t] = wmma_bf16(fq[1].v, fk[t][1].v, z);
    }

    // ---- online softmax (fully unrolled: constant vector subscripts) ----
#pragma unroll
    for (int r = 0; r < 8; ++r) {
      float s0 = sc[0][r], s1 = sc[1][r];
      float mx = fmaxf(s0, s1);
      mx = fmaxf(mx, __shfl_xor(mx, 1, 32));
      mx = fmaxf(mx, __shfl_xor(mx, 2, 32));
      mx = fmaxf(mx, __shfl_xor(mx, 4, 32));
      mx = fmaxf(mx, __shfl_xor(mx, 8, 32));
      float mnew  = fmaxf(mrow[r], mx);
      float alpha = __expf(mrow[r] - mnew);
      mrow[r] = mnew;
      float p0 = __expf(s0 - mnew);
      float p1 = __expf(s1 - mnew);
      oL[r] *= alpha;
#pragma unroll
      for (int t = 0; t < 4; ++t) o[t][r] *= alpha;
      Plds[wave][r + half * 8][l15]      = (bf16_t)p0;
      Plds[wave][r + half * 8][16 + l15] = (bf16_t)p1;
    }
    // wave-private LDS RAW: wait the CDNA5 split DS counter
    asm volatile("s_wait_dscnt 0" ::: "memory");

    // P as A-operand (16 q x 32 keys)
    FragBF fp;
    fp.q[0] = *(const u32x4*)(&Plds[wave][l15][koff]);
    fp.q[1] = *(const u32x4*)(&Plds[wave][l15][16 + koff]);

    // ---- O += P * V ; row-sum l += P * ones via one extra WMMA ----
#pragma unroll
    for (int t = 0; t < 4; ++t)
      o[t] = wmma_bf16(fp.v, fv[t].v, o[t]);
    oL = wmma_bf16(fp.v, fone.v, oL);
  }

  // normalize and store to att[B,S,D] (bf16); every column of oL holds l[row]
#pragma unroll
  for (int t = 0; t < 4; ++t)
#pragma unroll
    for (int r = 0; r < 8; ++r) {
      int qrow = q0 + r + half * 8;
      float v = o[t][r] / oL[r];
      att[((size_t)b * SS + qrow) * DD + h * DH + t * 16 + l15] = (bf16_t)v;
    }
}

// ---------------------------------------------------------------------------
extern "C" void kernel_launch(void* const* d_in, const int* in_sizes, int n_in,
                              void* d_out, int out_size, void* d_ws, size_t ws_size,
                              hipStream_t stream) {
  const float* x     = (const float*)d_in[0];
  const float* w_in  = (const float*)d_in[1];
  const float* b_in  = (const float*)d_in[2];
  const float* w_out = (const float*)d_in[3];
  const float* b_out = (const float*)d_in[4];
  float* out = (float*)d_out;

  char* ws = (char*)d_ws;
  size_t off = 0;
  auto carve = [&](size_t bytes) -> void* {
    void* p = ws + off;
    off += (bytes + 255) & ~(size_t)255;
    return p;
  };
  const size_t nBSD = (size_t)BB * SS * DD;        // 8,388,608
  bf16_t* xb    = (bf16_t*)carve(nBSD * 2);
  bf16_t* w_inb = (bf16_t*)carve((size_t)D3 * DD * 2);
  bf16_t* w_oub = (bf16_t*)carve((size_t)DD * DD * 2);
  bf16_t* Qb    = (bf16_t*)carve(nBSD * 2);
  bf16_t* Kb    = (bf16_t*)carve(nBSD * 2);
  bf16_t* Vtb   = (bf16_t*)carve(nBSD * 2);
  bf16_t* attb  = (bf16_t*)carve(nBSD * 2);

  // 1) convert inputs/weights to bf16
  {
    int n4 = (int)(nBSD / 4);
    cvt_f32_bf16_kernel<<<n4 / 256, 256, 0, stream>>>(x, xb, n4);
    n4 = D3 * DD / 4;
    cvt_f32_bf16_kernel<<<n4 / 256, 256, 0, stream>>>(w_in, w_inb, n4);
    n4 = DD * DD / 4;
    cvt_f32_bf16_kernel<<<n4 / 256, 256, 0, stream>>>(w_out, w_oub, n4);
  }

  // 2) QKV projection (M=B*S=8192, N=3072, K=1024) -> scatter Q/K/Vt
  gemm_bf16_nt<0><<<dim3((BB * SS) / 128, D3 / 128), 256, 0, stream>>>(
      xb, w_inb, b_in, nullptr, Qb, Kb, Vtb, BB * SS, D3, DD);

  // 3) flash attention over 64 (b,h) pairs x 16 q-tiles
  flash_attn_kernel<<<dim3(SS / 128, BB * HH), 256, 0, stream>>>(Qb, Kb, Vtb, attb);

  // 4) output projection (M=8192, N=1024, K=1024) -> f32 out + b_out
  gemm_bf16_nt<1><<<dim3((BB * SS) / 128, DD / 128), 256, 0, stream>>>(
      attb, w_oub, b_out, out, nullptr, nullptr, nullptr, BB * SS, DD, DD);
}
